// SillyLinear_52578989638264
// MI455X (gfx1250) — matrix operand order repaired and checked by reading, not compiled, not run on hardware
//
#include <hip/hip_runtime.h>
#include <hip/hip_bf16.h>

// ---------------------------------------------------------------------------
// SillyLinear: W = sum_d coef[d]*w_basis[d], b = sum_d bcoef[d]*b_basis[d],
//              y = x @ W^T + b
// Shapes (fixed): x[16384, 2048], w_basis[64, 2048, 2048], y[16384, 2048]
// Strategy: stream-reduce basis -> W (HBM-bound, ~46us floor at 23.3 TB/s),
//           stage x/W to f16 in workspace (both fit in 192MB L2), then
//           v_wmma_f32_16x16x32_f16 GEMM with 64x64 wave tiles.
// ---------------------------------------------------------------------------

typedef __attribute__((ext_vector_type(16))) _Float16 v16h;
typedef __attribute__((ext_vector_type(8)))  _Float16 v8h;
typedef __attribute__((ext_vector_type(4)))  _Float16 v4h;
typedef __attribute__((ext_vector_type(8)))  float    v8f;

#define M_DIM   16384      // B*S
#define N_DIM   2048       // OUT_F
#define K_DIM   2048       // IN_F
#define D_DIM   64         // INT_DIM

// ---------------------------------------------------------------------------
// Kernel 1: convert x f32 -> f16 (vectorized, 4 elems/thread)
// ---------------------------------------------------------------------------
__global__ __launch_bounds__(256) void xf16_kernel(
    const float* __restrict__ x, _Float16* __restrict__ xh)
{
    size_t i = (size_t)blockIdx.x * blockDim.x + threadIdx.x;   // float4 index
    const float4 v = ((const float4*)x)[i];
    v4h h;
    h[0] = (_Float16)v.x; h[1] = (_Float16)v.y;
    h[2] = (_Float16)v.z; h[3] = (_Float16)v.w;
    ((v4h*)xh)[i] = h;
}

// ---------------------------------------------------------------------------
// Kernel 2: W[o,i] = sum_d coef[d] * w_basis[d,o,i]  -> f16
// Streams 1 GiB of basis once; each thread owns 4 consecutive W elements.
// ---------------------------------------------------------------------------
__global__ __launch_bounds__(256) void buildw_kernel(
    const float* __restrict__ coef,
    const float* __restrict__ basis,     // [64, 2048*2048]
    _Float16* __restrict__ Wh)           // [2048*2048] f16
{
    const size_t idx4    = (size_t)blockIdx.x * blockDim.x + threadIdx.x;
    const size_t stride4 = (size_t)N_DIM * K_DIM / 4;     // float4 per basis slab
    const float4* b4 = (const float4*)basis;

    float ax = 0.f, ay = 0.f, az = 0.f, aw = 0.f;
#pragma unroll 4
    for (int d = 0; d < D_DIM; ++d) {
        const float  c = coef[d];                  // uniform -> scalar load
        const float4 v = b4[(size_t)d * stride4 + idx4];
        ax = fmaf(c, v.x, ax);
        ay = fmaf(c, v.y, ay);
        az = fmaf(c, v.z, az);
        aw = fmaf(c, v.w, aw);
    }
    v4h h;
    h[0] = (_Float16)ax; h[1] = (_Float16)ay;
    h[2] = (_Float16)az; h[3] = (_Float16)aw;
    ((v4h*)Wh)[idx4] = h;
}

// ---------------------------------------------------------------------------
// Kernel 3: bias[o] = sum_d bcoef[d] * b_basis[d,o]
// ---------------------------------------------------------------------------
__global__ __launch_bounds__(256) void buildb_kernel(
    const float* __restrict__ bcoef,
    const float* __restrict__ bbasis,    // [64, 2048]
    float* __restrict__ bias)            // [2048]
{
    const int o = blockIdx.x * blockDim.x + threadIdx.x;
    if (o >= N_DIM) return;
    float acc = 0.f;
#pragma unroll 8
    for (int d = 0; d < D_DIM; ++d)
        acc = fmaf(bcoef[d], bbasis[(size_t)d * N_DIM + o], acc);
    bias[o] = acc;
}

// ---------------------------------------------------------------------------
// Kernel 4: WMMA GEMM  C[m,n] = sum_k A[m,k]*Bw[n,k] + bias[n]
//   A  : x_f16  [M, K] row-major
//   Bw : W_f16  [N, K] row-major  (== B matrix column data, K-contiguous)
// Block tile 256x128, 8 waves (4 along M x 2 along N), wave tile 64x64:
//   16 WMMAs per K=32 step vs 16 b128 loads (~32 flop per L2 byte).
// Bias is folded into accumulator initialization (col = lane&15 per tile).
// ---------------------------------------------------------------------------
__global__ __launch_bounds__(256) void gemm_wmma_kernel(
    const _Float16* __restrict__ A,
    const _Float16* __restrict__ Bw,
    const float*    __restrict__ bias,
    float* __restrict__ C)
{
    const int wave  = threadIdx.x >> 5;
    const int lane  = threadIdx.x & 31;
    const int lr    = lane & 15;          // row/col within tile
    const int lh    = (lane >> 4) & 1;    // lane half select

    const int m0 = blockIdx.y * 256 + (wave & 3) * 64;   // wave's first row
    const int n0 = blockIdx.x * 128 + (wave >> 2) * 64;  // wave's first col

    // A fragment base pointers: lane holds row (m0+mi*16+lr),
    // chunk0 at k0 + lh*16, chunk1 at k0 + lh*16 + 8.
    const _Float16* aPtr[4];
#pragma unroll
    for (int mi = 0; mi < 4; ++mi)
        aPtr[mi] = A + (size_t)(m0 + mi * 16 + lr) * K_DIM + lh * 16;

    // B fragment base pointers: lane holds column (n0+ni*16+lr),
    // 16 contiguous K values at k0 + lh*16.
    const _Float16* bPtr[4];
#pragma unroll
    for (int ni = 0; ni < 4; ++ni)
        bPtr[ni] = Bw + (size_t)(n0 + ni * 16 + lr) * K_DIM + lh * 16;

    // Initialize accumulators with bias (C/D layout: col = lane&15, all rows
    // in a lane share the same column within a tile).
    v8f acc[4][4];
#pragma unroll
    for (int ni = 0; ni < 4; ++ni) {
        const float bb = bias[n0 + ni * 16 + lr];
#pragma unroll
        for (int mi = 0; mi < 4; ++mi)
#pragma unroll
            for (int v = 0; v < 8; ++v)
                acc[mi][ni][v] = bb;
    }

    for (int k0 = 0; k0 < K_DIM; k0 += 32) {
        // Unconditional speculative prefetch of the next K slice into the
        // near cache (global_prefetch_b8); invalid tail addresses are dropped.
        __builtin_prefetch(aPtr[0] + k0 + 32, 0, 3);
        __builtin_prefetch(bPtr[0] + k0 + 32, 0, 3);

        v16h a[4];
#pragma unroll
        for (int mi = 0; mi < 4; ++mi) {
            const v8h lo = *(const v8h*)(aPtr[mi] + k0);
            const v8h hi = *(const v8h*)(aPtr[mi] + k0 + 8);
            a[mi] = __builtin_shufflevector(lo, hi,
                        0, 1, 2, 3, 4, 5, 6, 7, 8, 9, 10, 11, 12, 13, 14, 15);
        }

        v16h b[4];
#pragma unroll
        for (int ni = 0; ni < 4; ++ni)
            b[ni] = *(const v16h*)(bPtr[ni] + k0);

#pragma unroll
        for (int mi = 0; mi < 4; ++mi)
#pragma unroll
            for (int ni = 0; ni < 4; ++ni)
                acc[mi][ni] = __builtin_amdgcn_wmma_f32_16x16x32_f16(
                    /*neg_a=*/false, a[mi],
                    /*neg_b=*/false, b[ni],
                    /*c_mod=*/(short)0, acc[mi][ni],
                    /*reuse_a=*/false, /*reuse_b=*/false);
    }

    // Store: C/D layout -> VGPR v, lane L: row = v + 8*(L>=16), col = L&15
#pragma unroll
    for (int ni = 0; ni < 4; ++ni) {
        const int col = n0 + ni * 16 + lr;
#pragma unroll
        for (int mi = 0; mi < 4; ++mi) {
            const int rowBase = m0 + mi * 16 + lh * 8;
#pragma unroll
            for (int v = 0; v < 8; ++v)
                C[(size_t)(rowBase + v) * N_DIM + col] = acc[mi][ni][v];
        }
    }
}

// ---------------------------------------------------------------------------
extern "C" void kernel_launch(void* const* d_in, const int* in_sizes, int n_in,
                              void* d_out, int out_size, void* d_ws, size_t ws_size,
                              hipStream_t stream) {
    (void)in_sizes; (void)n_in; (void)out_size; (void)ws_size;

    const float* x      = (const float*)d_in[0];   // [16384, 2048]
    const float* wcoef  = (const float*)d_in[1];   // [64]
    const float* bcoef  = (const float*)d_in[2];   // [64]
    const float* wbasis = (const float*)d_in[3];   // [64, 2048, 2048]
    const float* bbasis = (const float*)d_in[4];   // [64, 2048]
    float* out = (float*)d_out;                    // [16384, 2048]

    char* ws = (char*)d_ws;
    _Float16* x_h  = (_Float16*)ws;                                   // 64 MiB
    _Float16* W_h  = (_Float16*)(ws + (size_t)64 * 1024 * 1024);      //  8 MiB
    float*    bias = (float*)   (ws + (size_t)72 * 1024 * 1024);      //  8 KiB

    // 1) x -> f16  (16384*2048/4 = 8,388,608 float4s)
    xf16_kernel<<<(M_DIM * (size_t)K_DIM) / 4 / 256, 256, 0, stream>>>(x, x_h);

    // 2) W = coef . basis  (2048*2048/4 = 1,048,576 float4s)
    buildw_kernel<<<(N_DIM * (size_t)K_DIM) / 4 / 256, 256, 0, stream>>>(
        wcoef, wbasis, W_h);

    // 3) bias
    buildb_kernel<<<N_DIM / 256, 256, 0, stream>>>(bcoef, bbasis, bias);

    // 4) GEMM: grid (N/128, M/256) = (16, 64), 256 threads = 8 waves
    gemm_wmma_kernel<<<dim3(N_DIM / 128, M_DIM / 256), 256, 0, stream>>>(
        x_h, W_h, bias, out);
}
